// StandardAttention_35811437314648
// MI455X (gfx1250) — matrix operand order repaired and checked by reading, not compiled, not run on hardware
//
#include <hip/hip_runtime.h>

// Problem constants (match reference)
#define BB     2
#define MM     2048
#define NN     2048
#define DMODEL 1024
#define NHEAD  16
#define DHEAD  64

typedef _Float16 v16h  __attribute__((ext_vector_type(16)));
typedef _Float16 v8h   __attribute__((ext_vector_type(8)));
typedef float    v8f   __attribute__((ext_vector_type(8)));
typedef float    f32x8 __attribute__((ext_vector_type(8)));

union Frag { v16h v; v8h h[2]; };

static __device__ __forceinline__ v8f wmma_f16(const Frag& a, const Frag& b, v8f c) {
  // D = A(16x32 f16) * B(32x16 f16) + C(16x16 f32)
  return __builtin_amdgcn_wmma_f32_16x16x32_f16(
      /*neg_a=*/false, a.v, /*neg_b=*/false, b.v,
      /*c_mod=*/(short)0, c, /*reuse_a=*/false, /*reuse_b=*/false);
}

// ---------------------------------------------------------------------------
// 16-lane row max using DPP butterflies (pure VALU, no LDS pipe traffic).
// ---------------------------------------------------------------------------
template <int CTRL>
static __device__ __forceinline__ float dpp_fmax(float x) {
  int y = __builtin_amdgcn_update_dpp(0, __float_as_int(x), CTRL, 0xf, 0xf, true);
  return fmaxf(x, __int_as_float(y));
}
static __device__ __forceinline__ float rowmax16(float x) {
  x = dpp_fmax<0xB1>(x);   // quad_perm(1,0,3,2)
  x = dpp_fmax<0x4E>(x);   // quad_perm(2,3,0,1)
  x = dpp_fmax<0x141>(x);  // row_half_mirror
  x = dpp_fmax<0x140>(x);  // row_mirror
  return x;
}

// ---------------------------------------------------------------------------
// LDS-tiled convert+transpose: fp32 W (K=1024, N=1024) -> f16 Wt (N, K).
// Both global sides coalesced.
// ---------------------------------------------------------------------------
__global__ void __launch_bounds__(256) wcvt_kernel(const float* __restrict__ w,
                                                   _Float16* __restrict__ wt) {
  __shared__ _Float16 tile[32][33];
  const int kt = blockIdx.x & 31;          // K/32
  const int nt = (blockIdx.x >> 5) & 31;   // N/32
  const int x = threadIdx.x & 31;
  const int y = threadIdx.x >> 5;          // 0..7
#pragma unroll
  for (int q = 0; q < 4; ++q) {
    const int row = y * 4 + q;
    tile[row][x] = (_Float16)w[(size_t)(kt * 32 + row) * DMODEL + nt * 32 + x];
  }
  __syncthreads();
#pragma unroll
  for (int q = 0; q < 4; ++q) {
    const int row = y * 4 + q;
    wt[(size_t)(nt * 32 + row) * DMODEL + kt * 32 + x] = tile[x][row];
  }
}

// ---------------------------------------------------------------------------
// Vectorized fp32 -> f16 activation conversion (8 elems/thread).
// ---------------------------------------------------------------------------
__global__ void __launch_bounds__(256) acvt_kernel(const float* __restrict__ x,
                                                   _Float16* __restrict__ y) {
  const int idx = (blockIdx.x * 256 + threadIdx.x) * 8;
  f32x8 t = *(const f32x8*)(x + idx);
  v8h o;
#pragma unroll
  for (int e = 0; e < 8; ++e) o[e] = (_Float16)t[e];
  *(v8h*)(y + idx) = o;
}

// ---------------------------------------------------------------------------
// Tiled WMMA GEMM: C(4096 x 1024) = A(4096 x 1024) * Bt^T
//   A: f16 row-major. Bt: f16 (1024 x 1024) stored N-major (n,k).
//   Each wave computes a 32x64 tile: 2x4 accumulators, K-step 32.
// ---------------------------------------------------------------------------
template <bool OUT16>
__global__ void __launch_bounds__(256) gemm_kernel(const _Float16* __restrict__ A,
                                                   const _Float16* __restrict__ Bt,
                                                   void* __restrict__ Cout) {
  const int lane = threadIdx.x & 31;
  const int wave = threadIdx.x >> 5;
  const int l15  = lane & 15;
  const int half = lane >> 4;
  const int lin  = blockIdx.x * 8 + wave;   // 2048 wave-tasks total
  const int m0   = (lin >> 4) * 32;
  const int n0   = (lin & 15) * 64;

  v8f acc[2][4] = {};

  for (int k0 = 0; k0 < DMODEL; k0 += 32) {
    Frag a[2];
#pragma unroll
    for (int i = 0; i < 2; ++i) {
      const _Float16* ap = A + (size_t)(m0 + 16 * i + l15) * DMODEL + k0 + 8 * half;
      a[i].h[0] = *(const v8h*)ap;
      a[i].h[1] = *(const v8h*)(ap + 16);
      if (k0 + 32 < DMODEL) __builtin_prefetch((const void*)(ap + 32), 0, 1);
    }
#pragma unroll
    for (int j = 0; j < 4; ++j) {
      Frag b;
      const _Float16* bp = Bt + (size_t)(n0 + 16 * j + l15) * DMODEL + k0 + 8 * half;
      b.h[0] = *(const v8h*)bp;
      b.h[1] = *(const v8h*)(bp + 16);
#pragma unroll
      for (int i = 0; i < 2; ++i) acc[i][j] = wmma_f16(a[i], b, acc[i][j]);
    }
    if (k0 + 32 < DMODEL)
      __builtin_prefetch((const void*)(Bt + (size_t)(n0 + l15) * DMODEL + k0 + 32), 0, 1);
  }

#pragma unroll
  for (int i = 0; i < 2; ++i)
#pragma unroll
    for (int j = 0; j < 4; ++j) {
      const int col = n0 + 16 * j + l15;
#pragma unroll
      for (int r = 0; r < 8; ++r) {
        const int row = m0 + 16 * i + r + 8 * half;
        const float v = acc[i][j][r];
        if constexpr (OUT16)
          ((_Float16*)Cout)[(size_t)row * DMODEL + col] = (_Float16)v;
        else
          ((float*)Cout)[(size_t)row * DMODEL + col] = v;
      }
    }
}

// ---------------------------------------------------------------------------
// LDS-tiled transpose: Vh (B, N, 1024) f16 -> Vt (B, 1024, N) f16.
// ---------------------------------------------------------------------------
__global__ void __launch_bounds__(256) vtrans_kernel(const _Float16* __restrict__ Vh,
                                                     _Float16* __restrict__ Vt) {
  __shared__ _Float16 tile[32][33];
  const int nt = blockIdx.x & 63;          // N/32
  const int ht = (blockIdx.x >> 6) & 31;   // 1024/32
  const int b  = blockIdx.x >> 11;
  const int x = threadIdx.x & 31;
  const int y = threadIdx.x >> 5;
#pragma unroll
  for (int k = 0; k < 4; ++k) {
    const int row = y * 4 + k;
    tile[row][x] = Vh[(size_t)(b * NN + nt * 32 + row) * DMODEL + ht * 32 + x];
  }
  __syncthreads();
#pragma unroll
  for (int k = 0; k < 4; ++k) {
    const int row = y * 4 + k;
    Vt[(size_t)(b * DMODEL + ht * 32 + row) * NN + nt * 32 + x] = tile[x][row];
  }
}

// ---------------------------------------------------------------------------
// Flash attention: one wave per (b, h, 32-row query tile), 64-key chunks.
// Per chunk: 16 WMMA for S, DPP row-max, exp2 softmax, P->LDS relayout,
// 4 WMMA row-sums (P x ones), 16 WMMA for O += P*V.
// ---------------------------------------------------------------------------
__global__ void __launch_bounds__(256) attn_kernel(const _Float16* __restrict__ Qh,
                                                   const _Float16* __restrict__ Kh,
                                                   const _Float16* __restrict__ Vt,
                                                   const unsigned char* __restrict__ mask,
                                                   _Float16* __restrict__ AO) {
  __shared__ __align__(16) _Float16 plds[8][32][80];   // per-wave P staging (40 KB)

  const int lane = threadIdx.x & 31;
  const int wave = threadIdx.x >> 5;
  const int l15  = lane & 15;
  const int half = lane >> 4;
  const int lin  = blockIdx.x * 8 + wave;   // 2048 wave-tasks exactly
  const int mt   = lin & 63;                // M/32
  const int h    = (lin >> 6) & 15;
  const int b    = lin >> 10;
  const int m0   = mt * 32;
  const float scale2 = 0.125f * 1.4426950408889634f;  // (1/sqrt(dh)) * log2(e)
  const float NEG_INF = -__builtin_inff();

  // Q fragments: 2 row subtiles x 2 k-chunks of DH=64
  Frag qa[2][2];
#pragma unroll
  for (int i = 0; i < 2; ++i) {
    const _Float16* qp = Qh + (size_t)(b * MM + m0 + 16 * i + l15) * DMODEL + h * DHEAD;
#pragma unroll
    for (int c = 0; c < 2; ++c) {
      qa[i][c].h[0] = *(const v8h*)(qp + 32 * c + 8 * half);
      qa[i][c].h[1] = *(const v8h*)(qp + 32 * c + 16 + 8 * half);
    }
  }

  Frag onef;
#pragma unroll
  for (int e = 0; e < 16; ++e) onef.v[e] = (_Float16)1.f;

  float mx[2][8];
#pragma unroll
  for (int i = 0; i < 2; ++i)
#pragma unroll
    for (int r = 0; r < 8; ++r) mx[i][r] = NEG_INF;
  v8f smacc[2] = {};   // running softmax denominators (C-layout rows)
  v8f acc[2][4] = {};  // O accumulators

  const unsigned char* mrow = mask + (size_t)b * NN;
  const _Float16* kbase = Kh + ((size_t)b * NN + l15) * DMODEL + h * DHEAD;
  const _Float16* vbase = Vt + ((size_t)b * DMODEL + h * DHEAD + l15) * NN;

  for (int n0 = 0; n0 < NN; n0 += 64) {
    bool mk[4];
#pragma unroll
    for (int t = 0; t < 4; ++t) mk[t] = mrow[n0 + 16 * t + l15] != 0;

    // Scores: 2 row subtiles x 4 key tiles; K fragments shared across subtiles
    v8f s[2][4] = {};
#pragma unroll
    for (int t = 0; t < 4; ++t) {
      const _Float16* kp = kbase + (size_t)(n0 + 16 * t) * DMODEL;
#pragma unroll
      for (int c = 0; c < 2; ++c) {
        Frag kb;
        kb.h[0] = *(const v8h*)(kp + 32 * c + 8 * half);
        kb.h[1] = *(const v8h*)(kp + 32 * c + 8 * half + 16);
#pragma unroll
        for (int i = 0; i < 2; ++i) s[i][t] = wmma_f16(qa[i][c], kb, s[i][t]);
      }
    }
    if (n0 + 64 < NN) {
#pragma unroll
      for (int t = 0; t < 4; ++t)
        __builtin_prefetch((const void*)(kbase + (size_t)(n0 + 64 + 16 * t) * DMODEL), 0, 1);
    }

    __builtin_amdgcn_wave_barrier();
    // Online softmax (base-2 domain); row = 16*i + r + 8*half
#pragma unroll
    for (int i = 0; i < 2; ++i)
#pragma unroll
      for (int r = 0; r < 8; ++r) {
        float v[4];
#pragma unroll
        for (int t = 0; t < 4; ++t) v[t] = mk[t] ? NEG_INF : s[i][t][r] * scale2;
        float cm = fmaxf(fmaxf(v[0], v[1]), fmaxf(v[2], v[3]));
        cm = rowmax16(cm);
        const float nm = fmaxf(mx[i][r], cm);
        const float alpha = (nm == NEG_INF) ? 1.f : __builtin_amdgcn_exp2f(mx[i][r] - nm);
        mx[i][r] = nm;
        smacc[i][r] *= alpha;
#pragma unroll
        for (int j = 0; j < 4; ++j) acc[i][j][r] *= alpha;
#pragma unroll
        for (int t = 0; t < 4; ++t) {
          const float p = mk[t] ? 0.f : __builtin_amdgcn_exp2f(v[t] - nm);
          plds[wave][16 * i + r + 8 * half][16 * t + l15] = (_Float16)p;
        }
      }
    __builtin_amdgcn_wave_barrier();

    // Re-read P as A-fragments: 2 subtiles x 2 k-chunks of the 64-key window
    Frag pa[2][2];
#pragma unroll
    for (int i = 0; i < 2; ++i)
#pragma unroll
      for (int pk = 0; pk < 2; ++pk) {
        pa[i][pk].h[0] = *(const v8h*)&plds[wave][16 * i + l15][32 * pk + 8 * half];
        pa[i][pk].h[1] = *(const v8h*)&plds[wave][16 * i + l15][32 * pk + 16 + 8 * half];
      }
    __builtin_amdgcn_wave_barrier();

    // Row sums via WMMA: smacc += P x ones
#pragma unroll
    for (int i = 0; i < 2; ++i)
#pragma unroll
      for (int pk = 0; pk < 2; ++pk) smacc[i] = wmma_f16(pa[i][pk], onef, smacc[i]);

    // O += P * V (V fragments shared across the two row subtiles)
#pragma unroll
    for (int j = 0; j < 4; ++j)
#pragma unroll
      for (int pk = 0; pk < 2; ++pk) {
        const _Float16* vp = vbase + (size_t)(16 * j) * NN + n0 + 32 * pk + 8 * half;
        Frag vb;
        vb.h[0] = *(const v8h*)vp;
        vb.h[1] = *(const v8h*)(vp + 16);
#pragma unroll
        for (int i = 0; i < 2; ++i) acc[i][j] = wmma_f16(pa[i][pk], vb, acc[i][j]);
      }
  }

  // Normalize and write concat-head output (f16)
#pragma unroll
  for (int i = 0; i < 2; ++i)
#pragma unroll
    for (int r = 0; r < 8; ++r) {
      const float inv = 1.f / smacc[i][r];
      const int row = m0 + 16 * i + r + 8 * half;
      _Float16* op = AO + (size_t)(b * MM + row) * DMODEL + h * DHEAD;
#pragma unroll
      for (int j = 0; j < 4; ++j) op[16 * j + l15] = (_Float16)(acc[i][j][r] * inv);
    }
}

// ---------------------------------------------------------------------------
// Launch
// ---------------------------------------------------------------------------
extern "C" void kernel_launch(void* const* d_in, const int* in_sizes, int n_in,
                              void* d_out, int out_size, void* d_ws, size_t ws_size,
                              hipStream_t stream) {
  const float* query     = (const float*)d_in[0];
  const float* key_value = (const float*)d_in[1];
  const unsigned char* kpm = (const unsigned char*)d_in[2];   // bool mask (1 byte/elem)
  const float* W_Q = (const float*)d_in[3];
  const float* W_K = (const float*)d_in[4];
  const float* W_V = (const float*)d_in[5];
  const float* W_O = (const float*)d_in[6];

  _Float16* ws = (_Float16*)d_ws;
  const size_t MEG = 1024 * 1024;
  _Float16* wq_t = ws + 0 * MEG;   // 1M f16 each
  _Float16* wk_t = ws + 1 * MEG;
  _Float16* wv_t = ws + 2 * MEG;
  _Float16* wo_t = ws + 3 * MEG;
  _Float16* q16  = ws + 4 * MEG;   // 4M f16 each
  _Float16* kv16 = ws + 8 * MEG;
  _Float16* Qh   = ws + 12 * MEG;
  _Float16* Kh   = ws + 16 * MEG;
  _Float16* Vh   = ws + 20 * MEG;
  _Float16* Vt   = ws + 24 * MEG;  // (B, 1024, N)
  _Float16* AO   = ws + 28 * MEG;  // attention output, concat heads

  // Weights: fp32 -> f16, transposed to N-major (LDS-tiled, coalesced)
  wcvt_kernel<<<1024, 256, 0, stream>>>(W_Q, wq_t);
  wcvt_kernel<<<1024, 256, 0, stream>>>(W_K, wk_t);
  wcvt_kernel<<<1024, 256, 0, stream>>>(W_V, wv_t);
  wcvt_kernel<<<1024, 256, 0, stream>>>(W_O, wo_t);

  // Activations: fp32 -> f16 once (kv reused by K and V projections)
  acvt_kernel<<<2048, 256, 0, stream>>>(query, q16);
  acvt_kernel<<<2048, 256, 0, stream>>>(key_value, kv16);

  // Projections (all-f16 GEMM, f16 out)
  gemm_kernel<true><<<256, 256, 0, stream>>>(q16, wq_t, Qh);
  gemm_kernel<true><<<256, 256, 0, stream>>>(kv16, wk_t, Kh);
  gemm_kernel<true><<<256, 256, 0, stream>>>(kv16, wv_t, Vh);

  // V -> (B, 1024, N), LDS-tiled
  vtrans_kernel<<<4096, 256, 0, stream>>>(Vh, Vt);

  // Attention (2048 waves, 32 query rows x 64-key chunks)
  attn_kernel<<<256, 256, 0, stream>>>(Qh, Kh, Vt, kpm, AO);

  // Output projection (f16 activations, fp32 out)
  gemm_kernel<false><<<256, 256, 0, stream>>>(AO, wo_t, (float*)d_out);
}
